// Encoder_GIN_30013231464612
// MI455X (gfx1250) — matrix operand order repaired and checked by reading, not compiled, not run on hardware
//
#include <hip/hip_runtime.h>
#include <hip/hip_bf16.h>

#define DF      128     // feature dim
#define MTILE   64      // rows per block in MLP kernel
#define GRAPHS  256
#define OSTRIDE 384     // 3 layers * 128 concatenated

typedef __attribute__((ext_vector_type(16))) __bf16 v16bf;
typedef __attribute__((ext_vector_type(8)))  __bf16 v8bf;
typedef __attribute__((ext_vector_type(8)))  float  v8f;

// ---------------------------------------------------------------------------
// zero fill (float4 granularity; all sizes used are multiples of 4)
// ---------------------------------------------------------------------------
__global__ __launch_bounds__(256) void zero_f32_kernel(float* __restrict__ p, int n4) {
    int i = blockIdx.x * blockDim.x + threadIdx.x;
    if (i < n4) reinterpret_cast<float4*>(p)[i] = make_float4(0.f, 0.f, 0.f, 0.f);
}

// ---------------------------------------------------------------------------
// edge scatter: agg[dst] += h[src]; one wave (32 lanes) per edge, float4/lane.
// Native f32 atomics (relaxed, agent scope -> global_atomic_add_f32).
// h/agg both fit in the 192MB L2, so this phase is L2-resident.
// ---------------------------------------------------------------------------
__global__ __launch_bounds__(256) void scatter_kernel(
        const float* __restrict__ h, int hs,
        const int* __restrict__ src, const int* __restrict__ dst,
        float* __restrict__ agg, int E) {
    int t = blockIdx.x * blockDim.x + threadIdx.x;
    int e = t >> 5;
    if (e >= E) return;
    int lane = t & 31;
    int s = src[e], d = dst[e];
    float4 v = *reinterpret_cast<const float4*>(h + (size_t)s * hs + lane * 4);
    float* ap = agg + (size_t)d * DF + lane * 4;
    __hip_atomic_fetch_add(ap + 0, v.x, __ATOMIC_RELAXED, __HIP_MEMORY_SCOPE_AGENT);
    __hip_atomic_fetch_add(ap + 1, v.y, __ATOMIC_RELAXED, __HIP_MEMORY_SCOPE_AGENT);
    __hip_atomic_fetch_add(ap + 2, v.z, __ATOMIC_RELAXED, __HIP_MEMORY_SCOPE_AGENT);
    __hip_atomic_fetch_add(ap + 3, v.w, __ATOMIC_RELAXED, __HIP_MEMORY_SCOPE_AGENT);
}

// ---------------------------------------------------------------------------
// One wave computes a 16x128 output stripe as 8 WMMA 16x16 tiles, K=128 in
// 4 steps of 32 (bf16 A/B, f32 acc). sA: [m][k] row-major bf16. sW: [n][k]
// (transposed) bf16 so B fragments are two 16B LDS vector loads each.
// ISA 16-bit layouts: lanes 0-15 hold k 0-7,16-23, lanes 16-31 hold k 8-15,24-31.
// ---------------------------------------------------------------------------
__device__ inline void gemm_stripe(const __bf16* __restrict__ sA,
                                   const __bf16* __restrict__ sW,
                                   int aRowIdx, int l15, int half, v8f acc[8]) {
    const __bf16* aRow = sA + aRowIdx * DF;
#pragma unroll
    for (int kb = 0; kb < DF; kb += 32) {
        v8bf alo = *reinterpret_cast<const v8bf*>(aRow + kb + half * 8);
        v8bf ahi = *reinterpret_cast<const v8bf*>(aRow + kb + 16 + half * 8);
        v16bf a = __builtin_shufflevector(alo, ahi,
                    0, 1, 2, 3, 4, 5, 6, 7, 8, 9, 10, 11, 12, 13, 14, 15);
#pragma unroll
        for (int nt = 0; nt < 8; nt++) {
            v8bf blo = *reinterpret_cast<const v8bf*>(
                sW + (nt * 16 + l15) * DF + kb + half * 8);
            v8bf bhi = *reinterpret_cast<const v8bf*>(
                sW + (nt * 16 + l15) * DF + kb + 16 + half * 8);
            v16bf b = __builtin_shufflevector(blo, bhi,
                    0, 1, 2, 3, 4, 5, 6, 7, 8, 9, 10, 11, 12, 13, 14, 15);
            acc[nt] = __builtin_amdgcn_wmma_f32_16x16x32_bf16(
                false, a, false, b, (short)0, acc[nt], false, false);
        }
    }
}

// ---------------------------------------------------------------------------
// Fused GIN MLP: z = h + agg; y = relu(z@W1+b1); h' = relu(y@W2+b2);
// store h' into xs slice and atomically pool into pooled slice.
// Block: 128 threads = 4 waves; 64 rows x 128 cols per block.
// ---------------------------------------------------------------------------
__global__ __launch_bounds__(128) void gin_mlp_kernel(
        const float* __restrict__ hprev, int hs,
        const float* __restrict__ agg,
        const float* __restrict__ W1, const float* __restrict__ b1,
        const float* __restrict__ W2, const float* __restrict__ b2,
        float* __restrict__ hout,          // base of this layer's xs slice
        const int* __restrict__ batch,
        float* __restrict__ pooled,        // base of this layer's pooled slice
        int N) {
    __shared__ __align__(32) __bf16 sW[DF * DF];     // W transposed [n][k]
    __shared__ __align__(32) __bf16 sA[MTILE * DF];  // activations  [m][k]

    const int tid  = threadIdx.x;
    const int lane = tid & 31;
    const int wave = tid >> 5;
    const int l15  = lane & 15;
    const int half = lane >> 4;
    const int row0 = blockIdx.x * MTILE;

    // Stage A = h + agg as bf16 (zero-pad rows past N)
    for (int idx = tid; idx < MTILE * (DF / 4); idx += 128) {
        int m = idx / (DF / 4);
        int c = (idx % (DF / 4)) * 4;
        int gm = row0 + m;
        float4 hv = make_float4(0.f, 0.f, 0.f, 0.f);
        float4 av = make_float4(0.f, 0.f, 0.f, 0.f);
        if (gm < N) {
            hv = *reinterpret_cast<const float4*>(hprev + (size_t)gm * hs + c);
            av = *reinterpret_cast<const float4*>(agg + (size_t)gm * DF + c);
        }
        sA[m * DF + c + 0] = (__bf16)(hv.x + av.x);
        sA[m * DF + c + 1] = (__bf16)(hv.y + av.y);
        sA[m * DF + c + 2] = (__bf16)(hv.z + av.z);
        sA[m * DF + c + 3] = (__bf16)(hv.w + av.w);
    }
    // Stage W1 transposed into LDS as bf16
    for (int idx = tid; idx < DF * DF; idx += 128) {
        int k = idx >> 7, n = idx & 127;
        sW[n * DF + k] = (__bf16)W1[idx];   // W1 is row-major [k][n]
    }
    __syncthreads();

    // GEMM 1
    v8f acc[8] = {};
    gemm_stripe(sA, sW, wave * 16 + l15, l15, half, acc);

    // bias + relu
#pragma unroll
    for (int nt = 0; nt < 8; nt++) {
        float bv = b1[nt * 16 + l15];
#pragma unroll
        for (int r = 0; r < 8; r++)
            acc[nt][r] = fmaxf(acc[nt][r] + bv, 0.f);
    }

    __syncthreads();   // everyone done reading sA/sW before overwrite

    // Re-stage Y1 -> sA (bf16) using the C/D layout (lane<16: M=r, N=l15;
    // lane>=16: M=r+8) and stage W2 transposed -> sW.
#pragma unroll
    for (int nt = 0; nt < 8; nt++)
#pragma unroll
        for (int r = 0; r < 8; r++)
            sA[(wave * 16 + half * 8 + r) * DF + nt * 16 + l15] = (__bf16)acc[nt][r];
    for (int idx = tid; idx < DF * DF; idx += 128) {
        int k = idx >> 7, n = idx & 127;
        sW[n * DF + k] = (__bf16)W2[idx];
    }
    __syncthreads();

    // GEMM 2
    v8f acc2[8] = {};
    gemm_stripe(sA, sW, wave * 16 + l15, l15, half, acc2);

    // bias + relu + store xs slice + pooled atomics
#pragma unroll
    for (int nt = 0; nt < 8; nt++) {
        float bv = b2[nt * 16 + l15];
        int col = nt * 16 + l15;
#pragma unroll
        for (int r = 0; r < 8; r++) {
            int gm = row0 + wave * 16 + half * 8 + r;
            if (gm < N) {
                float v = fmaxf(acc2[nt][r] + bv, 0.f);
                hout[(size_t)gm * OSTRIDE + col] = v;
                __hip_atomic_fetch_add(
                    pooled + (size_t)batch[gm] * OSTRIDE + col, v,
                    __ATOMIC_RELAXED, __HIP_MEMORY_SCOPE_AGENT);
            }
        }
    }
}

// ---------------------------------------------------------------------------
// launch: pooled region of d_out zeroed, then per layer:
//   zero agg -> scatter edges -> fused MLP (+pool). xs region of d_out is
//   also the storage for h between layers (stride 384, col offset 128*i).
// ---------------------------------------------------------------------------
extern "C" void kernel_launch(void* const* d_in, const int* in_sizes, int n_in,
                              void* d_out, int out_size, void* d_ws, size_t ws_size,
                              hipStream_t stream) {
    const float* x     = (const float*)d_in[0];
    const int*   ei    = (const int*)d_in[1];   // [2,E] (harness-normalized ints)
    const int*   batch = (const int*)d_in[2];   // [N]
    const float* W1    = (const float*)d_in[3]; // [3,128,128]
    const float* b1    = (const float*)d_in[4]; // [3,128]
    const float* W2    = (const float*)d_in[5];
    const float* b2    = (const float*)d_in[6];

    const int N = in_sizes[0] / DF;
    const int E = in_sizes[1] / 2;
    const int* src = ei;
    const int* dst = ei + E;

    float* out    = (float*)d_out;
    float* pooled = out;                          // [256, 384]
    float* xs     = out + (size_t)GRAPHS * OSTRIDE; // [N, 384]
    float* agg    = (float*)d_ws;                 // [N, 128] scratch

    {   // zero pooled output region (harness poisons d_out)
        int n4 = GRAPHS * OSTRIDE / 4;
        zero_f32_kernel<<<(n4 + 255) / 256, 256, 0, stream>>>(pooled, n4);
    }

    for (int i = 0; i < 3; i++) {
        const float* hp = (i == 0) ? x : (const float*)(xs + 128 * (i - 1));
        int hs          = (i == 0) ? DF : OSTRIDE;

        {   // zero agg
            int n4 = N * DF / 4;
            zero_f32_kernel<<<(n4 + 255) / 256, 256, 0, stream>>>(agg, n4);
        }
        {   // scatter-add over edges (one wave per edge)
            long long T = (long long)E * 32;
            scatter_kernel<<<(int)((T + 255) / 256), 256, 0, stream>>>(
                hp, hs, src, dst, agg, E);
        }
        gin_mlp_kernel<<<(N + MTILE - 1) / MTILE, 128, 0, stream>>>(
            hp, hs, agg,
            W1 + (size_t)i * DF * DF, b1 + (size_t)i * DF,
            W2 + (size_t)i * DF * DF, b2 + (size_t)i * DF,
            xs + 128 * i, batch, pooled + 128 * i, N);
    }
}
